// ConvDiscriminator_74586402063124
// MI455X (gfx1250) — matrix operand order repaired and checked by reading, not compiled, not run on hardware
//
#include <hip/hip_runtime.h>
#include <hip/hip_bf16.h>
#include <math.h>

// Problem constants (from reference): T=512, B=128, H=512, NF=128, FS=5
#define T_DIM 512
#define B_DIM 128
#define H_DIM 512
#define NFILT 128
#define FSZ   5
#define NPAIR 15              // sum_{w=1..5} w
#define ROWS (NPAIR * NFILT)  // 1920 rows in the stacked GEMM
#define CHUNK 16              // batches per chunk (bounds scratch to ~71 MB)
#define NCHUNK (B_DIM / CHUNK)

typedef __attribute__((ext_vector_type(16))) _Float16 v16h;
typedef __attribute__((ext_vector_type(8)))  _Float16 v8h;
typedef __attribute__((ext_vector_type(8)))  float    v8f;

// ---------------------------------------------------------------------------
// Kernel 1: stack conv weights into Wall[1920][512] f16, row r = 128*g + f,
// where g enumerates (w,dw) pairs in order w=1..5, dw=0..w-1 (prefix 0,1,3,6,10).
// Wall[r][h] = conv_w{w}[f,0,h,dw]   (K = h contiguous -> A fragment friendly)
// ---------------------------------------------------------------------------
__global__ void pack_weights_kernel(const float* __restrict__ w1,
                                    const float* __restrict__ w2,
                                    const float* __restrict__ w3,
                                    const float* __restrict__ w4,
                                    const float* __restrict__ w5,
                                    _Float16* __restrict__ Wall) {
    int tid = blockIdx.x * blockDim.x + threadIdx.x;
    if (tid >= ROWS * H_DIM) return;
    int h = tid & (H_DIM - 1);
    int r = tid >> 9;            // / 512
    int f = r & (NFILT - 1);
    int g = r >> 7;              // / 128 -> (w,dw) pair index 0..14
    int w, dw; const float* src;
    if      (g < 1)  { w = 1; dw = g;      src = w1; }
    else if (g < 3)  { w = 2; dw = g - 1;  src = w2; }
    else if (g < 6)  { w = 3; dw = g - 3;  src = w3; }
    else if (g < 10) { w = 4; dw = g - 6;  src = w4; }
    else             { w = 5; dw = g - 10; src = w5; }
    float v = src[(size_t)f * H_DIM * w + (size_t)h * w + dw];
    Wall[(size_t)r * H_DIM + h] = (_Float16)v;
}

// ---------------------------------------------------------------------------
// Kernel 2: per chunk, materialize the "raw reshape" gather, TRANSPOSED:
//   Xt[bl][t][h] = (t < L) ? enc[(h*L+t)/H, b, (h*L+t)%H] : 0     (f16)
// t-major layout => B fragments are contiguous-K 32B loads.
// ---------------------------------------------------------------------------
__global__ void build_x_kernel(const float* __restrict__ enc,
                               const int* __restrict__ lengths,
                               _Float16* __restrict__ Xt, int bBase) {
    int tid = blockIdx.x * 256 + threadIdx.x;
    int h  = tid & (H_DIM - 1);
    int t  = (tid >> 9) & (T_DIM - 1);
    int bl = tid >> 18;
    int b  = bBase + bl;
    int L  = lengths[b];
    float v = 0.0f;
    if (t < L) {
        int idx = h * L + t;                       // < T*H
        v = enc[(size_t)(idx >> 9) * (B_DIM * H_DIM) + (size_t)b * H_DIM + (idx & 511)];
    }
    Xt[(size_t)bl * T_DIM * H_DIM + (size_t)t * H_DIM + h] = (_Float16)v;
}

// ---------------------------------------------------------------------------
// Kernel 3: GEMM  Y[bl][r][t] = sum_h Wall[r][h] * Xt[bl][t][h]   (f32 accum)
// 1920 x 512 x 512 per batch via v_wmma_f32_16x16x32_f16.
// Block = 128 threads = 4 waves. Each wave owns a 16(row) x 64(col) strip:
// one A fragment against 4 B fragments per K-step (4 WMMAs / 10 b128 loads).
// Software pipeline with TWO register fragment buffers: the next K-tile's
// loads are always in flight while the current tile's WMMAs execute, so the
// wait before a WMMA group is partial (loadcnt<=6), not 0.
// Fragment layouts used here (contiguous K, same order in A and B):
//   A (16x32 f16): lane<16 holds row=lane, K = [k, k+16);
//                  lane>=16 holds row=lane-16, K = [k+16, k+32).
//   B (32x16 f16): lane<16 holds col=lane,   K = [k, k+16);
//                  lane>=16 holds col=lane-16, K = [k+16, k+32).
//   C/D f32: VGPR j <-> row j + 8*(lane>=16), col = lane&15.
// ---------------------------------------------------------------------------
__global__ void __launch_bounds__(128)
gemm_chunk_kernel(const _Float16* __restrict__ Wall,
                  const _Float16* __restrict__ Xt,
                  float* __restrict__ Y) {
    const int lane = threadIdx.x & 31;
    const int wv   = threadIdx.x >> 5;
    const int t0   = blockIdx.x * 64;                 // col block: 4 tiles of 16
    const int r0   = (blockIdx.y * 4 + wv) * 16;      // row tile (per wave)
    const int bl   = blockIdx.z;                      // batch within chunk

    const int selA = (lane < 16) ? 0 : 16;
    const int selB = (lane < 16) ? 0 : 16;
    const int lcol = lane & 15;

    const _Float16* aPtr = Wall + (size_t)(r0 + lcol) * H_DIM + selA;
    const _Float16* bPtr = Xt + (size_t)bl * T_DIM * H_DIM
                              + (size_t)(t0 + lcol) * H_DIM + selB;

    union AF { v16h v; v8h h[2]; };
    AF   A[2];
    v16h Bf[2][4];
    v8f  c0 = {}, c1 = {}, c2 = {}, c3 = {};

    auto loadFrags = [&](int slot, int kk) {
        A[slot].h[0] = *(const v8h*)(aPtr + kk);         // K sel..sel+7
        A[slot].h[1] = *(const v8h*)(aPtr + kk + 8);     // K sel+8..sel+15
        Bf[slot][0]  = *(const v16h*)(bPtr + kk);
        Bf[slot][1]  = *(const v16h*)(bPtr + (size_t)16 * H_DIM + kk);
        Bf[slot][2]  = *(const v16h*)(bPtr + (size_t)32 * H_DIM + kk);
        Bf[slot][3]  = *(const v16h*)(bPtr + (size_t)48 * H_DIM + kk);
    };
    auto doWmma = [&](int slot) {
        c0 = __builtin_amdgcn_wmma_f32_16x16x32_f16(false, A[slot].v, false,
                 Bf[slot][0], (short)0, c0, false, false);
        c1 = __builtin_amdgcn_wmma_f32_16x16x32_f16(false, A[slot].v, false,
                 Bf[slot][1], (short)0, c1, false, false);
        c2 = __builtin_amdgcn_wmma_f32_16x16x32_f16(false, A[slot].v, false,
                 Bf[slot][2], (short)0, c2, false, false);
        c3 = __builtin_amdgcn_wmma_f32_16x16x32_f16(false, A[slot].v, false,
                 Bf[slot][3], (short)0, c3, false, false);
    };

    loadFrags(0, 0);                                   // prologue: tile 0
#pragma unroll 1
    for (int k = 0; k < H_DIM - 64; k += 64) {         // 7 iterations
        loadFrags(1, k + 32);                          // next tile in flight
        __builtin_prefetch(aPtr + k + 64, 0, 1);
        __builtin_prefetch(bPtr + k + 64, 0, 1);
        doWmma(0);                                     // compute tile k/32
        loadFrags(0, k + 64);
        doWmma(1);                                     // compute tile k/32+1
    }
    loadFrags(1, H_DIM - 32);                          // tile 15
    doWmma(0);                                         // tile 14
    doWmma(1);                                         // tile 15

    const int rowHalf = (lane < 16) ? 0 : 8;
    float* y = Y + ((size_t)bl * ROWS + r0 + rowHalf) * T_DIM + t0 + lcol;
#pragma unroll
    for (int j = 0; j < 8; ++j) {
        y[(size_t)j * T_DIM +  0] = c0[j];
        y[(size_t)j * T_DIM + 16] = c1[j];
        y[(size_t)j * T_DIM + 32] = c2[j];
        y[(size_t)j * T_DIM + 48] = c3[j];
    }
}

// ---------------------------------------------------------------------------
// Kernel 4: epilogue. One wave per (bl, w, f):
//   m = max(0, max_{t<=Leff-w} ( bias_w[f] + sum_dw Y[bl][128*(pref+dw)+f][t+dw] ))
// (ReLU + zero-mask + max over t collapse to max with 0.)
// Wave shuffle max-reduce, lane 0 writes feat[b][(w-1)*128+f].
// ---------------------------------------------------------------------------
__global__ void epilogue_kernel(const float* __restrict__ Y,
                                const int* __restrict__ lengths,
                                const float* __restrict__ b1, const float* __restrict__ b2,
                                const float* __restrict__ b3, const float* __restrict__ b4,
                                const float* __restrict__ b5,
                                float* __restrict__ feat, int bBase) {
    const int wid  = blockIdx.x * 4 + (threadIdx.x >> 5);
    const int lane = threadIdx.x & 31;
    const int f    = wid & (NFILT - 1);
    const int wm1  = (wid >> 7) % FSZ;
    const int bl   = wid / (NFILT * FSZ);
    const int w    = wm1 + 1;
    const int b    = bBase + bl;

    const int prefA[FSZ] = {0, 1, 3, 6, 10};
    const float* biasArr[FSZ] = {b1, b2, b3, b4, b5};
    const float bias = biasArr[wm1][f];

    int L    = lengths[b];
    int Leff = (L > FSZ) ? L : FSZ;
    int tMax = Leff - w;                      // inclusive last valid t

    const float* base = Y + ((size_t)bl * ROWS + (size_t)NFILT * prefA[wm1] + f) * T_DIM;

    float m = 0.0f;
    for (int t = lane; t <= tMax; t += 32) {
        float s = bias;
        for (int dw = 0; dw < w; ++dw)
            s += base[(size_t)dw * NFILT * T_DIM + t + dw];
        m = fmaxf(m, s);
    }
    for (int off = 16; off; off >>= 1)
        m = fmaxf(m, __shfl_xor(m, off, 32));
    if (lane == 0)
        feat[(size_t)b * (FSZ * NFILT) + wm1 * NFILT + f] = m;
}

// ---------------------------------------------------------------------------
// Kernel 5: tiny FC head: feat(128x640) -> fc1(100) -> fc2(1) -> sigmoid.
// ---------------------------------------------------------------------------
__global__ void __launch_bounds__(128)
fc_head_kernel(const float* __restrict__ feat,
               const float* __restrict__ fc1w, const float* __restrict__ fc1b,
               const float* __restrict__ fc2w, const float* __restrict__ fc2b,
               float* __restrict__ out) {
    __shared__ float s1[100];
    const int b = blockIdx.x;
    const int j = threadIdx.x;
    const float* fv = feat + (size_t)b * (FSZ * NFILT);
    if (j < 100) {
        float acc = fc1b[j];
        const float* wrow = fc1w + (size_t)j * (FSZ * NFILT);
        for (int i = 0; i < FSZ * NFILT; ++i) acc += fv[i] * wrow[i];
        s1[j] = acc;
    }
    __syncthreads();
    if (j == 0) {
        float acc = fc2b[0];
        for (int i = 0; i < 100; ++i) acc += s1[i] * fc2w[i];
        out[b] = 1.0f / (1.0f + expf(-acc));
    }
}

// ---------------------------------------------------------------------------
extern "C" void kernel_launch(void* const* d_in, const int* in_sizes, int n_in,
                              void* d_out, int out_size, void* d_ws, size_t ws_size,
                              hipStream_t stream) {
    const float* enc     = (const float*)d_in[0];
    const int*   lengths = (const int*)d_in[1];
    const float* cw[FSZ] = {(const float*)d_in[2], (const float*)d_in[4],
                            (const float*)d_in[6], (const float*)d_in[8],
                            (const float*)d_in[10]};
    const float* cb[FSZ] = {(const float*)d_in[3], (const float*)d_in[5],
                            (const float*)d_in[7], (const float*)d_in[9],
                            (const float*)d_in[11]};
    const float* fc1w = (const float*)d_in[12];
    const float* fc1b = (const float*)d_in[13];
    const float* fc2w = (const float*)d_in[14];
    const float* fc2b = (const float*)d_in[15];

    // Scratch layout (all 256B-aligned by construction): ~71 MB total.
    char* ws = (char*)d_ws;
    size_t offW = 0;                                            // 1.9 MB
    size_t offX = offW + (size_t)ROWS * H_DIM * 2;              // 8 MB
    size_t offY = offX + (size_t)CHUNK * T_DIM * H_DIM * 2;     // 60 MB
    size_t offF = offY + (size_t)CHUNK * ROWS * T_DIM * 4;      // 0.3 MB
    _Float16* Wall = (_Float16*)(ws + offW);
    _Float16* Xt   = (_Float16*)(ws + offX);
    float*    Y    = (float*)(ws + offY);
    float*    feat = (float*)(ws + offF);

    pack_weights_kernel<<<(ROWS * H_DIM) / 256, 256, 0, stream>>>(
        cw[0], cw[1], cw[2], cw[3], cw[4], Wall);

    for (int c = 0; c < NCHUNK; ++c) {
        const int bBase = c * CHUNK;
        build_x_kernel<<<(CHUNK * T_DIM * H_DIM) / 256, 256, 0, stream>>>(
            enc, lengths, Xt, bBase);
        dim3 g(T_DIM / 64, ROWS / 64, CHUNK);   // (8, 30, 16)
        gemm_chunk_kernel<<<g, 128, 0, stream>>>(Wall, Xt, Y);
        epilogue_kernel<<<(CHUNK * FSZ * NFILT) / 4, 128, 0, stream>>>(
            Y, lengths, cb[0], cb[1], cb[2], cb[3], cb[4], feat, bBase);
    }

    fc_head_kernel<<<B_DIM, 128, 0, stream>>>(feat, fc1w, fc1b, fc2w, fc2b,
                                              (float*)d_out);
}
